// MultiHeadAttention_12841952215822
// MI455X (gfx1250) — matrix operand order repaired and checked by reading, not compiled, not run on hardware
//
#include <hip/hip_runtime.h>

// ---------------------------------------------------------------------------
// MHA forward on MI455X (gfx1250, wave32, WMMA).
// Compute-bound (~69 GFLOP vs ~100MB HBM traffic) -> all GEMMs on
// v_wmma_f32_16x16x32_bf16 (f32 accumulate). bf16 staging buffers in d_ws.
// GEMMs: 128x128 block tiles, A/B panels double-buffered in LDS via
// global_load_async_to_lds_b128 (ASYNCcnt) to cut L2 traffic ~3x.
// Flash-attention: wave-shared running max + row sums via all-ones WMMA tile.
// ---------------------------------------------------------------------------

typedef __attribute__((ext_vector_type(16))) __bf16 bf16x16;
typedef __attribute__((ext_vector_type(8)))  float  f32x8;

static constexpr int Bc = 2;
static constexpr int Tc = 2048;
static constexpr int Cc = 1024;
static constexpr int Hc = 16;
static constexpr int Dc = 64;
static constexpr int M_TOK = Bc * Tc;       // 4096 token rows

union FragBF { bf16x16 v; uint4 q[2]; };

static __device__ __forceinline__ unsigned short f2bf(float f) {
  unsigned u = __float_as_uint(f);
  u += 0x7fffu + ((u >> 16) & 1u);          // round-to-nearest-even
  return (unsigned short)(u >> 16);
}

static __device__ __forceinline__ f32x8 wmma_bf(const FragBF& a, const FragBF& b, f32x8 c) {
  // (neg_a, A, neg_b, B, c_mod, C, reuse_a, reuse_b)
  return __builtin_amdgcn_wmma_f32_16x16x32_bf16(false, a.v, false, b.v, (short)0, c, false, false);
}

// Async global->LDS copy (16B per lane), tracked by ASYNCcnt.
static __device__ __forceinline__ void async_cp16(unsigned lds_addr, const void* gaddr) {
  asm volatile("global_load_async_to_lds_b128 %0, %1, off"
               :: "v"(lds_addr), "v"(gaddr) : "memory");
}
static __device__ __forceinline__ void wait_async0() {
#if __has_builtin(__builtin_amdgcn_s_wait_asynccnt)
  __builtin_amdgcn_s_wait_asynccnt(0);
#else
  asm volatile("s_wait_asynccnt 0x0" ::: "memory");
#endif
}

// ---------------- 1. fp32 -> bf16 elementwise ------------------------------
__global__ void k_cvt_bf16(const float* __restrict__ in,
                           unsigned short* __restrict__ out, int n) {
  int i = blockIdx.x * blockDim.x + threadIdx.x;
  if (i < n) out[i] = f2bf(in[i]);
}

// ---------------- 2. W (K x N) fp32 -> Wt (N x K) bf16 ---------------------
__global__ void k_transpose_w(const float* __restrict__ in,
                              unsigned short* __restrict__ out, int K, int N) {
  __shared__ float tile[32][33];
  int n0 = blockIdx.x * 32, k0 = blockIdx.y * 32;
  int tx = threadIdx.x, ty = threadIdx.y;           // (32, 8)
#pragma unroll
  for (int j = 0; j < 4; ++j)
    tile[ty + j * 8][tx] = in[(size_t)(k0 + ty + j * 8) * N + n0 + tx];
  __syncthreads();
#pragma unroll
  for (int j = 0; j < 4; ++j)
    out[(size_t)(n0 + ty + j * 8) * K + k0 + tx] = f2bf(tile[tx][ty + j * 8]);
}

// ---------------- 3. V slice of qkv -> Vt[bh][d][t] bf16 -------------------
__global__ void k_pack_vt(const unsigned short* __restrict__ qkv,
                          unsigned short* __restrict__ vt) {
  __shared__ unsigned short tile[32][33];
  int bh = blockIdx.z, b = bh / Hc, h = bh % Hc;
  int t0 = blockIdx.x * 32, d0 = blockIdx.y * 32;
  int tx = threadIdx.x, ty = threadIdx.y;           // (32, 8)
#pragma unroll
  for (int j = 0; j < 4; ++j) {
    int t = t0 + ty + j * 8;
    tile[ty + j * 8][tx] =
        qkv[(size_t)(b * Tc + t) * (3 * Cc) + 2 * Cc + h * Dc + d0 + tx];
  }
  __syncthreads();
#pragma unroll
  for (int j = 0; j < 4; ++j) {
    int d = d0 + ty + j * 8;
    vt[((size_t)bh * Dc + d) * Tc + t0 + tx] = tile[tx][ty + j * 8];
  }
}

// ---------------- 4. bf16 GEMM: Out = A(MxK) * Bt(NxK)^T + bias ------------
// 256 threads = 8 waves (4M x 2N), block tile 128x128, wave tile 32x64.
// A/B 128x32 panels double-buffered in LDS via async copies.
static constexpr int BK   = 32;    // k-step (halfs)
static constexpr int PADR = 40;    // LDS row pitch in halfs (80B, 16B aligned)

template <bool OUT_BF16>
__global__ void __launch_bounds__(256, 1)
k_gemm_bf16(const unsigned short* __restrict__ A,
            const unsigned short* __restrict__ Bt,
            const float* __restrict__ bias,
            void* __restrict__ Out, int M, int N, int K) {
  __shared__ __align__(16) unsigned short lA[2][128 * PADR];   // 2 x 10KB
  __shared__ __align__(16) unsigned short lB[2][128 * PADR];   // 2 x 10KB
  const int tid  = threadIdx.x;
  const int lane = tid & 31;
  const int w    = tid >> 5;            // 0..7
  const int wm   = w & 3, wn = w >> 2;  // 4 x 2 wave grid
  const int mblk = blockIdx.y * 128, nblk = blockIdx.x * 128;
  const int r    = lane & 15, hi = lane >> 4;

  // each thread copies 2 x 16B chunks of A and of B per k-step (512 chunks)
  const int c0   = tid * 2;
  const int row0 = c0 >> 2,      off0 = (c0 & 3) * 8;       // halfs
  const int row1 = (c0 + 1) >> 2, off1 = ((c0 + 1) & 3) * 8;

  auto issue = [&](int buf, int k0) {
    async_cp16((unsigned)(size_t)&lA[buf][row0 * PADR + off0],
               A + (size_t)(mblk + row0) * K + k0 + off0);
    async_cp16((unsigned)(size_t)&lA[buf][row1 * PADR + off1],
               A + (size_t)(mblk + row1) * K + k0 + off1);
    async_cp16((unsigned)(size_t)&lB[buf][row0 * PADR + off0],
               Bt + (size_t)(nblk + row0) * K + k0 + off0);
    async_cp16((unsigned)(size_t)&lB[buf][row1 * PADR + off1],
               Bt + (size_t)(nblk + row1) * K + k0 + off1);
  };

  f32x8 acc[2][4];
#pragma unroll
  for (int i = 0; i < 2; ++i)
#pragma unroll
    for (int j = 0; j < 4; ++j)
      acc[i][j] = (f32x8){0.f, 0.f, 0.f, 0.f, 0.f, 0.f, 0.f, 0.f};

  issue(0, 0);
  int buf = 0;
  for (int k0 = 0; k0 < K; k0 += BK) {
    wait_async0();            // my copies for buf are done
    __syncthreads();          // -> everyone's are; prev reads of buf^1 retired
    if (k0 + BK < K) issue(buf ^ 1, k0 + BK);

    FragBF a0, a1;
    {
      const unsigned short* pa = &lA[buf][(wm * 32 + r) * PADR + hi * 8];
      a0.q[0] = *(const uint4*)pa;        a0.q[1] = *(const uint4*)(pa + 16);
      pa += 16 * PADR;
      a1.q[0] = *(const uint4*)pa;        a1.q[1] = *(const uint4*)(pa + 16);
    }
#pragma unroll
    for (int nf = 0; nf < 4; ++nf) {
      FragBF bf;
      const unsigned short* pb = &lB[buf][(wn * 64 + nf * 16 + r) * PADR + hi * 8];
      bf.q[0] = *(const uint4*)pb;        bf.q[1] = *(const uint4*)(pb + 16);
      acc[0][nf] = wmma_bf(a0, bf, acc[0][nf]);
      acc[1][nf] = wmma_bf(a1, bf, acc[1][nf]);
    }
    buf ^= 1;
  }
#pragma unroll
  for (int mi = 0; mi < 2; ++mi)
#pragma unroll
    for (int nf = 0; nf < 4; ++nf)
#pragma unroll
      for (int g = 0; g < 8; ++g) {
        int row = mblk + wm * 32 + mi * 16 + g + hi * 8;
        int col = nblk + wn * 64 + nf * 16 + r;
        float v = acc[mi][nf][g] + bias[col];
        if (OUT_BF16) ((unsigned short*)Out)[(size_t)row * N + col] = f2bf(v);
        else          ((float*)Out)[(size_t)row * N + col] = v;
      }
}

// ---------------- 5. Flash attention: one wave = (b,h, 16 queries) ---------
__global__ void __launch_bounds__(128, 1)
k_attn(const unsigned short* __restrict__ qkv,
       const unsigned short* __restrict__ vt,
       const int* __restrict__ amask,
       unsigned short* __restrict__ O) {
  __shared__ __align__(16) unsigned short pbuf[4][16 * 32];  // per-wave P tile
  const int lane  = threadIdx.x & 31;
  const int wslot = threadIdx.x >> 5;
  const int wave  = (blockIdx.x * blockDim.x + threadIdx.x) >> 5;
  const int mt    = Tc / 16;
  const int mtile = wave % mt;
  const int bh    = wave / mt;
  const int b = bh / Hc, h = bh % Hc;
  const int m0 = mtile * 16;
  const int r = lane & 15, hi = lane >> 4;
  const float scale = 0.03125f;                     // C^-0.5 (matches reference)

  FragBF qf[2];
  {
    const unsigned short* p = qkv + (size_t)(b * Tc + m0 + r) * (3 * Cc) + h * Dc;
#pragma unroll
    for (int kq = 0; kq < 2; ++kq) {
      const unsigned short* pp = p + kq * 32 + hi * 8;
      qf[kq].q[0] = *(const uint4*)pp;  qf[kq].q[1] = *(const uint4*)(pp + 16);
    }
  }
  FragBF ones;                                      // all-ones B tile (bf16 1.0)
  ones.q[0] = make_uint4(0x3F803F80u, 0x3F803F80u, 0x3F803F80u, 0x3F803F80u);
  ones.q[1] = ones.q[0];

  float m = -1e30f;                                 // wave-shared running max
  f32x8 Lacc = (f32x8){0.f, 0.f, 0.f, 0.f, 0.f, 0.f, 0.f, 0.f};
  f32x8 acc[4];
#pragma unroll
  for (int nf = 0; nf < 4; ++nf)
    acc[nf] = (f32x8){0.f, 0.f, 0.f, 0.f, 0.f, 0.f, 0.f, 0.f};

  for (int key0 = 0; key0 < m0 + 16; key0 += 32) {   // uniform per wave
    f32x8 S0 = (f32x8){0.f,0.f,0.f,0.f,0.f,0.f,0.f,0.f};
    f32x8 S1 = (f32x8){0.f,0.f,0.f,0.f,0.f,0.f,0.f,0.f};
    {
      const unsigned short* p =
          qkv + (size_t)(b * Tc + key0 + r) * (3 * Cc) + Cc + h * Dc;
      FragBF k0f, k1f;
      k0f.q[0] = *(const uint4*)(p + hi * 8);
      k0f.q[1] = *(const uint4*)(p + hi * 8 + 16);
      k1f.q[0] = *(const uint4*)(p + 32 + hi * 8);
      k1f.q[1] = *(const uint4*)(p + 32 + hi * 8 + 16);
      S0 = wmma_bf(qf[0], k0f, S0);
      S0 = wmma_bf(qf[1], k1f, S0);
    }
    {
      const unsigned short* p =
          qkv + (size_t)(b * Tc + key0 + 16 + r) * (3 * Cc) + Cc + h * Dc;
      FragBF k0f, k1f;
      k0f.q[0] = *(const uint4*)(p + hi * 8);
      k0f.q[1] = *(const uint4*)(p + hi * 8 + 16);
      k1f.q[0] = *(const uint4*)(p + 32 + hi * 8);
      k1f.q[1] = *(const uint4*)(p + 32 + hi * 8 + 16);
      S1 = wmma_bf(qf[0], k0f, S1);
      S1 = wmma_bf(qf[1], k1f, S1);
    }

    // ---- scale + mask + wave-wide max ----
    const int kc0 = key0 + r, kc1 = key0 + 16 + r;
    const int am0 = amask[b * Tc + kc0];
    const int am1 = amask[b * Tc + kc1];
    float mx = -1e30f;
#pragma unroll
    for (int g = 0; g < 8; ++g) {
      int qrow = m0 + g + hi * 8;
      float s0 = S0[g] * scale, s1 = S1[g] * scale;
      if (kc0 > qrow || am0 == 0) s0 = -1e30f;
      if (kc1 > qrow || am1 == 0) s1 = -1e30f;
      S0[g] = s0; S1[g] = s1;
      mx = fmaxf(mx, fmaxf(s0, s1));
    }
#pragma unroll
    for (int off = 1; off < 32; off <<= 1)           // 5 cross-lane maxes
      mx = fmaxf(mx, __shfl_xor(mx, off, 32));

    float mnew  = fmaxf(m, mx);
    float alpha = __expf(m - mnew);
    m = mnew;
    Lacc = Lacc * alpha;
#pragma unroll
    for (int nf = 0; nf < 4; ++nf) acc[nf] = acc[nf] * alpha;

    // ---- P = exp(S - m), written to LDS (C-layout -> A-layout) ----
#pragma unroll
    for (int g = 0; g < 8; ++g) {
      int prow = g + hi * 8;
      pbuf[wslot][prow * 32 + r]      = f2bf(__expf(S0[g] - mnew));
      pbuf[wslot][prow * 32 + 16 + r] = f2bf(__expf(S1[g] - mnew));
    }
    FragBF pf;                                       // intra-wave LDS in-order
    {
      const unsigned short* p = &pbuf[wslot][r * 32 + hi * 8];
      pf.q[0] = *(const uint4*)p;  pf.q[1] = *(const uint4*)(p + 16);
    }

    // ---- row sums via ones-column WMMA; O += P @ V ----
    Lacc = wmma_bf(pf, ones, Lacc);
#pragma unroll
    for (int nf = 0; nf < 4; ++nf) {
      FragBF vf;
      const unsigned short* p =
          vt + ((size_t)bh * Dc + nf * 16 + r) * Tc + key0 + hi * 8;
      vf.q[0] = *(const uint4*)p;  vf.q[1] = *(const uint4*)(p + 16);
      acc[nf] = wmma_bf(pf, vf, acc[nf]);
    }
  }

#pragma unroll
  for (int g = 0; g < 8; ++g) {
    float inv = 1.0f / Lacc[g];
    int row = m0 + g + hi * 8;
#pragma unroll
    for (int nf = 0; nf < 4; ++nf)
      O[(size_t)(b * Tc + row) * Cc + h * Dc + nf * 16 + r] = f2bf(acc[nf][g] * inv);
  }
}

// ---------------------------------------------------------------------------
extern "C" void kernel_launch(void* const* d_in, const int* in_sizes, int n_in,
                              void* d_out, int out_size, void* d_ws, size_t ws_size,
                              hipStream_t stream) {
  (void)in_sizes; (void)n_in; (void)out_size; (void)ws_size;
  const float* x      = (const float*)d_in[0];
  const int*   amask  = (const int*)d_in[1];
  const float* W_qkv  = (const float*)d_in[3];
  const float* b_qkv  = (const float*)d_in[4];
  const float* W_proj = (const float*)d_in[5];
  const float* b_proj = (const float*)d_in[6];

  char* ws = (char*)d_ws;
  size_t off = 0;
  unsigned short* Xbf    = (unsigned short*)(ws + off); off += (size_t)M_TOK * Cc * 2;        // 8 MB
  unsigned short* Wqkvt  = (unsigned short*)(ws + off); off += (size_t)3 * Cc * Cc * 2;       // 6 MB
  unsigned short* Wprojt = (unsigned short*)(ws + off); off += (size_t)Cc * Cc * 2;           // 2 MB
  unsigned short* qkv    = (unsigned short*)(ws + off); off += (size_t)M_TOK * 3 * Cc * 2;    // 24 MB
  unsigned short* Vt     = (unsigned short*)(ws + off); off += (size_t)Bc * Hc * Dc * Tc * 2; // 8 MB
  unsigned short* Obf    = (unsigned short*)(ws + off); off += (size_t)M_TOK * Cc * 2;        // 8 MB -> 56 MB

  // 1) X -> bf16
  k_cvt_bf16<<<(M_TOK * Cc) / 256, 256, 0, stream>>>(x, Xbf, M_TOK * Cc);
  // 2) transpose weights (fp32 -> bf16)
  k_transpose_w<<<dim3(3 * Cc / 32, Cc / 32), dim3(32, 8), 0, stream>>>(W_qkv, Wqkvt, Cc, 3 * Cc);
  k_transpose_w<<<dim3(Cc / 32, Cc / 32),     dim3(32, 8), 0, stream>>>(W_proj, Wprojt, Cc, Cc);
  // 3) qkv = X @ W_qkv + b_qkv   (M=4096, N=3072, K=1024) -> bf16
  k_gemm_bf16<true><<<dim3(3 * Cc / 128, M_TOK / 128), 256, 0, stream>>>(
      Xbf, Wqkvt, b_qkv, qkv, M_TOK, 3 * Cc, Cc);
  // 4) Vt[bh][d][t]
  k_pack_vt<<<dim3(Tc / 32, Dc / 32, Bc * Hc), dim3(32, 8), 0, stream>>>(qkv, Vt);
  // 5) flash attention -> Obf
  {
    int waves = Bc * Hc * (Tc / 16);                     // 4096
    k_attn<<<waves / 4, 128, 0, stream>>>(qkv, Vt, amask, Obf);
  }
  // 6) out = O @ W_proj + b_proj (M=4096, N=1024, K=1024) -> f32 d_out
  k_gemm_bf16<false><<<dim3(Cc / 128, M_TOK / 128), 256, 0, stream>>>(
      Obf, Wprojt, b_proj, d_out, M_TOK, Cc, Cc);
}
